// MultiHeadSelfAttention_33689723470278
// MI455X (gfx1250) — matrix operand order repaired and checked by reading, not compiled, not run on hardware
//
#include <hip/hip_runtime.h>
#include <hip/hip_bf16.h>

// ---------------------------------------------------------------------------
// MI455X (gfx1250) fused MHSA with sparse-mask softmax.
// All GEMMs on v_wmma_f32_16x16x32_bf16 (wave32 WMMA); scores stay in LDS.
//   - projection: 32x64 output per wave (2 A-frags x 4 B-frags -> 8 WMMA/step)
//   - attention : Q frags hoisted; probs stored bf16 in LDS for P@V A-frags
// ---------------------------------------------------------------------------

typedef __bf16 bf16_t;
typedef bf16_t v16bf __attribute__((ext_vector_type(16)));
typedef bf16_t v8bf  __attribute__((ext_vector_type(8)));
typedef bf16_t v4bf  __attribute__((ext_vector_type(4)));
typedef float  v8f   __attribute__((ext_vector_type(8)));

#define D_MODEL 1024
#define SEQ     1024
#define NB      4
#define NH      16
#define DK      64
#define TEMP_INV 100.0f      // 1/TEMP
#define THRESH   0.019f
#define NEG_INF  -1e9f
#define SCALE    0.125f      // 1/sqrt(64)

// ----- helpers --------------------------------------------------------------

__device__ __forceinline__ bf16_t f2bf(float f) {
  unsigned u = __builtin_bit_cast(unsigned, f);
  u += 0x7FFFu + ((u >> 16) & 1u);           // round-to-nearest-even
  unsigned short s = (unsigned short)(u >> 16);
  return __builtin_bit_cast(bf16_t, s);
}

__device__ __forceinline__ v16bf cat8(v8bf lo, v8bf hi) {
  return __builtin_shufflevector(lo, hi, 0,1,2,3,4,5,6,7,8,9,10,11,12,13,14,15);
}

// A fragment 16x32 (MxK), bf16, from row-major [ld] memory.
// ISA 7.12.2: lane m=lane&15, half=lane>>4; elems 0..7 -> K = half*8+i,
// elems 8..15 -> K = 16 + half*8 + (i-8). Two contiguous 16B chunks per lane.
__device__ __forceinline__ v16bf load_a_bf16(const bf16_t* p, int ld, int row0,
                                             int k0, int lane) {
  int m = lane & 15, hv = lane >> 4;
  const bf16_t* r = p + (long)(row0 + m) * ld + k0;
  v8bf lo = *(const v8bf*)(r + hv * 8);
  v8bf hi = *(const v8bf*)(r + 16 + hv * 8);
  return cat8(lo, hi);
}

// B fragment 32x16 (KxN), bf16, from N-major storage: mem[n][k], k contiguous.
// ISA: lane n=lane&15, half=lane>>4; elems 0..7 -> K = half*8+i, elems 8..15 -> K = 16 + half*8 + (i-8).
__device__ __forceinline__ v16bf load_b_bf16(const bf16_t* p, int ld, int n0,
                                             int k0, int lane) {
  int n = lane & 15, hv = lane >> 4;
  const bf16_t* r = p + (long)(n0 + n) * ld + k0;
  v8bf lo = *(const v8bf*)(r + hv * 8);
  v8bf hi = *(const v8bf*)(r + 16 + hv * 8);
  return cat8(lo, hi);
}

__device__ __forceinline__ v8f wmma_bf16(v16bf a, v16bf b, v8f c) {
  return __builtin_amdgcn_wmma_f32_16x16x32_bf16(false, a, false, b,
                                                 (short)0, c, false, false);
}

// ----- kernel 0: fp32 -> bf16 streaming convert (vectorized x4) -------------

__global__ __launch_bounds__(256)
void mhsa_cvt_bf16(const float4* __restrict__ in, v4bf* __restrict__ out, int n4) {
  int i = blockIdx.x * 256 + threadIdx.x;
  if (i < n4) {
    float4 f = in[i];
    v4bf o;
    o[0] = f2bf(f.x); o[1] = f2bf(f.y); o[2] = f2bf(f.z); o[3] = f2bf(f.w);
    out[i] = o;
  }
}

// ----- kernel 1: projection Y = X @ W^T + b, head-split store ---------------
// mode 0: q -> [B,H,S,Dk]   mode 1: k -> [B,H,S,Dk]   mode 2: v -> [B,H,Dk,S]
// One wave = one 32x64 output tile: per 32-wide K step,
//   2 A-frags + 4 B-frags feed 8 independent WMMAs (good load/math overlap).

__global__ __launch_bounds__(128)
void mhsa_proj(const bf16_t* __restrict__ X, const bf16_t* __restrict__ W,
               const float* __restrict__ bias, bf16_t* __restrict__ dst,
               int mode) {
  int wave = threadIdx.x >> 5;
  int lane = threadIdx.x & 31;
  int tile = blockIdx.x * 4 + wave;        // 0..2047
  int rt = tile >> 4;                      // row tile (0..127)   -> 32 rows of X
  int cg = tile & 15;                      // col group (0..15)   -> 64 output cols
  int row0 = rt * 32, o0 = cg * 64;

  v8f c00 = {}, c01 = {}, c02 = {}, c03 = {};
  v8f c10 = {}, c11 = {}, c12 = {}, c13 = {};
#pragma unroll 2
  for (int k0 = 0; k0 < D_MODEL; k0 += 32) {
    v16bf a0 = load_a_bf16(X, D_MODEL, row0,      k0, lane);
    v16bf a1 = load_a_bf16(X, D_MODEL, row0 + 16, k0, lane);
    v16bf b0 = load_b_bf16(W, D_MODEL, o0,      k0, lane);
    v16bf b1 = load_b_bf16(W, D_MODEL, o0 + 16, k0, lane);
    v16bf b2 = load_b_bf16(W, D_MODEL, o0 + 32, k0, lane);
    v16bf b3 = load_b_bf16(W, D_MODEL, o0 + 48, k0, lane);
    if (k0 + 32 < D_MODEL)
      __builtin_prefetch(X + (long)(row0 + (lane & 15)) * D_MODEL + k0 + 32, 0, 1);
    c00 = wmma_bf16(a0, b0, c00);
    c01 = wmma_bf16(a0, b1, c01);
    c02 = wmma_bf16(a0, b2, c02);
    c03 = wmma_bf16(a0, b3, c03);
    c10 = wmma_bf16(a1, b0, c10);
    c11 = wmma_bf16(a1, b1, c11);
    c12 = wmma_bf16(a1, b2, c12);
    c13 = wmma_bf16(a1, b3, c13);
  }

  int n = lane & 15, hv = lane >> 4;
  v8f cc[2][4] = {{c00, c01, c02, c03}, {c10, c11, c12, c13}};
#pragma unroll
  for (int rt2 = 0; rt2 < 2; ++rt2) {
#pragma unroll
    for (int t = 0; t < 4; ++t) {
      int col = o0 + t * 16 + n;
      float bvv = bias[col];
      int h = col >> 6, dk = col & 63;
#pragma unroll
      for (int r = 0; r < 8; ++r) {
        int grow = row0 + rt2 * 16 + hv * 8 + r;   // global row in [0, B*S)
        int b = grow >> 10, srow = grow & 1023;
        long offs;
        if (mode == 2)
          offs = (((long)b * NH + h) * DK + dk) * SEQ + srow;   // v transposed
        else
          offs = (((long)b * NH + h) * SEQ + srow) * DK + dk;
        dst[offs] = f2bf(cc[rt2][t][r] + bvv);
      }
    }
  }
}

// ----- kernel 2: attention for one (b,h, 16-query tile) ---------------------
// LDS: 16x1024 fp32 scores (64KB) + 16x1024 bf16 probs (32KB) = 96KB/WG.

__global__ __launch_bounds__(128)
void mhsa_attn(const bf16_t* __restrict__ qh, const bf16_t* __restrict__ kh,
               const bf16_t* __restrict__ vT, float* __restrict__ out) {
  __shared__ float  sc[16][SEQ];           // fp32 scores
  __shared__ bf16_t pb[16][SEQ];           // bf16 probabilities for P@V

  int blk  = blockIdx.x;                   // 0..4095
  int qt   = blk & 63;                     // query tile within head
  int bh   = blk >> 6;                     // b*NH + h
  int b    = bh >> 4, h = bh & 15;
  int lane = threadIdx.x & 31;
  int wave = threadIdx.x >> 5;             // 4 waves
  int row0 = qt * 16;

  const bf16_t* qb = qh + (long)bh * SEQ * DK;   // [S,Dk] row-major
  const bf16_t* kb = kh + (long)bh * SEQ * DK;   // [S,Dk] row-major (N-major B)
  const bf16_t* vb = vT + (long)bh * DK * SEQ;   // [Dk,S] row-major (N-major B)

  // ---- phase A: scores = (Q Kt)/8 into LDS; Q frags hoisted ---------------
  {
    v16bf aq0 = load_a_bf16(qb, DK, row0, 0,  lane);
    v16bf aq1 = load_a_bf16(qb, DK, row0, 32, lane);
#pragma unroll 2
    for (int kt = wave; kt < 64; kt += 4) {
      v16bf b0 = load_b_bf16(kb, DK, kt * 16, 0,  lane);
      v16bf b1 = load_b_bf16(kb, DK, kt * 16, 32, lane);
      v8f c = {};
      c = wmma_bf16(aq0, b0, c);
      c = wmma_bf16(aq1, b1, c);
      int n = lane & 15, hv = lane >> 4;
#pragma unroll
      for (int r = 0; r < 8; ++r)
        sc[hv * 8 + r][kt * 16 + n] = c[r] * SCALE;
    }
  }
  __syncthreads();

  // ---- phase B: sharp-softmax mask, masked softmax, bf16 prob store -------
  for (int row = wave; row < 16; row += 4) {
    float* srow = sc[row];

    float mx = -3.0e38f;
    for (int j = lane; j < SEQ; j += 32) mx = fmaxf(mx, srow[j]);
#pragma unroll
    for (int o = 16; o; o >>= 1) mx = fmaxf(mx, __shfl_xor(mx, o, 32));

    float sum = 0.0f;
    for (int j = lane; j < SEQ; j += 32) sum += __expf((srow[j] - mx) * TEMP_INV);
#pragma unroll
    for (int o = 16; o; o >>= 1) sum += __shfl_xor(sum, o, 32);
    float inv = 1.0f / sum;

    // mask: attn_sharp < THRESH  ->  score = NEG_INF
    for (int j = lane; j < SEQ; j += 32) {
      float s = srow[j];
      float p = __expf((s - mx) * TEMP_INV) * inv;
      srow[j] = (p < THRESH) ? NEG_INF : s;
    }

    float mx2 = -3.0e38f;
    for (int j = lane; j < SEQ; j += 32) mx2 = fmaxf(mx2, srow[j]);
#pragma unroll
    for (int o = 16; o; o >>= 1) mx2 = fmaxf(mx2, __shfl_xor(mx2, o, 32));

    float sum2 = 0.0f;
    for (int j = lane; j < SEQ; j += 32) sum2 += __expf(srow[j] - mx2);
#pragma unroll
    for (int o = 16; o; o >>= 1) sum2 += __shfl_xor(sum2, o, 32);
    float inv2 = 1.0f / sum2;

    for (int j = lane; j < SEQ; j += 32)
      pb[row][j] = f2bf(__expf(srow[j] - mx2) * inv2);
  }
  __syncthreads();

  // ---- phase C: ctx = P @ V ; wave w owns dk-tile w (16 cols) -------------
  {
    int nt = wave;                         // 0..3
    const bf16_t* pbase = &pb[0][0];       // [16,1024] bf16, row-major
    v8f c = {};
#pragma unroll 4
    for (int k0 = 0; k0 < SEQ; k0 += 32) {
      v16bf a  = load_a_bf16(pbase, SEQ, 0, k0, lane);       // ds_load_b128 x2
      v16bf bm = load_b_bf16(vb, SEQ, nt * 16, k0, lane);
      c = wmma_bf16(a, bm, c);
    }
    int n = lane & 15, hv = lane >> 4;
#pragma unroll
    for (int r = 0; r < 8; ++r) {
      int srowg = row0 + hv * 8 + r;
      out[((long)b * SEQ + srowg) * D_MODEL + h * DK + nt * 16 + n] = c[r];
    }
  }
}

// ----- host side -------------------------------------------------------------

extern "C" void kernel_launch(void* const* d_in, const int* in_sizes, int n_in,
                              void* d_out, int out_size, void* d_ws, size_t ws_size,
                              hipStream_t stream) {
  const float* Q  = (const float*)d_in[1];
  const float* K  = (const float*)d_in[2];
  const float* V  = (const float*)d_in[3];
  const float* Wq = (const float*)d_in[4];
  const float* bq = (const float*)d_in[5];
  const float* Wk = (const float*)d_in[6];
  const float* bk = (const float*)d_in[7];
  const float* Wv = (const float*)d_in[8];
  const float* bv = (const float*)d_in[9];
  float* out = (float*)d_out;

  const size_t NX = (size_t)NB * SEQ * D_MODEL;   // 4M elems
  const size_t NW = (size_t)D_MODEL * D_MODEL;    // 1M elems

  char* ws = (char*)d_ws;
  size_t off = 0;
  auto alloc = [&](size_t bytes) -> char* {
    char* p = ws + off;
    off = (off + bytes + 255) & ~(size_t)255;
    return p;
  };
  bf16_t* Qb  = (bf16_t*)alloc(NX * 2);
  bf16_t* Kb  = (bf16_t*)alloc(NX * 2);
  bf16_t* Vb  = (bf16_t*)alloc(NX * 2);
  bf16_t* Wqb = (bf16_t*)alloc(NW * 2);
  bf16_t* Wkb = (bf16_t*)alloc(NW * 2);
  bf16_t* Wvb = (bf16_t*)alloc(NW * 2);
  bf16_t* qhp = (bf16_t*)alloc(NX * 2);   // [B,H,S,Dk]
  bf16_t* khp = (bf16_t*)alloc(NX * 2);   // [B,H,S,Dk]
  bf16_t* vTp = (bf16_t*)alloc(NX * 2);   // [B,H,Dk,S]
  (void)ws_size;

  // convert inputs to bf16 (x4 vectorized)
  int nx4 = (int)(NX / 4), nw4 = (int)(NW / 4);
  mhsa_cvt_bf16<<<(nx4 + 255) / 256, 256, 0, stream>>>((const float4*)Q,  (v4bf*)Qb,  nx4);
  mhsa_cvt_bf16<<<(nx4 + 255) / 256, 256, 0, stream>>>((const float4*)K,  (v4bf*)Kb,  nx4);
  mhsa_cvt_bf16<<<(nx4 + 255) / 256, 256, 0, stream>>>((const float4*)V,  (v4bf*)Vb,  nx4);
  mhsa_cvt_bf16<<<(nw4 + 255) / 256, 256, 0, stream>>>((const float4*)Wq, (v4bf*)Wqb, nw4);
  mhsa_cvt_bf16<<<(nw4 + 255) / 256, 256, 0, stream>>>((const float4*)Wk, (v4bf*)Wkb, nw4);
  mhsa_cvt_bf16<<<(nw4 + 255) / 256, 256, 0, stream>>>((const float4*)Wv, (v4bf*)Wvb, nw4);

  // projections: 2048 wave-tiles (32x64 each), 4 waves/block -> 512 blocks
  mhsa_proj<<<512, 128, 0, stream>>>(Qb, Wqb, bq, qhp, 0);
  mhsa_proj<<<512, 128, 0, stream>>>(Kb, Wkb, bk, khp, 1);
  mhsa_proj<<<512, 128, 0, stream>>>(Vb, Wvb, bv, vTp, 2);

  // attention: B*H * (S/16) = 4096 blocks of 4 waves
  mhsa_attn<<<NB * NH * (SEQ / 16), 128, 0, stream>>>(qhp, khp, vTp, out);
  (void)in_sizes; (void)n_in; (void)out_size;
}
